// CDA_43731357008123
// MI455X (gfx1250) — matrix operand order repaired and checked by reading, not compiled, not run on hardware
//
#include <hip/hip_runtime.h>
#include <hip/hip_bf16.h>

// ---------------------------------------------------------------------------
// CDNA5 (gfx1250) implementation of the GCN + PCA + bilinear decoder pipeline.
// Dense linear algebra -> v_wmma_f32_16x16x32_bf16 (wave32 WMMA).
// W staging in the main GEMM -> Tensor Data Mover (tensor_load_to_lds) when
// the builtin is available, with a cooperative-copy fallback.
// ---------------------------------------------------------------------------

typedef __attribute__((ext_vector_type(16))) __bf16 v16bf;
typedef __attribute__((ext_vector_type(8)))  __bf16 v8bf;
typedef __attribute__((ext_vector_type(8)))  float  v8f;
typedef __attribute__((ext_vector_type(4)))  unsigned int u32x4;
typedef __attribute__((ext_vector_type(4)))  int i32x4;
typedef __attribute__((ext_vector_type(8)))  int i32x8;

#if __has_builtin(__builtin_amdgcn_tensor_load_to_lds) && \
    __has_builtin(__builtin_amdgcn_s_wait_tensorcnt)
#define HAVE_TDM 1
#else
#define HAVE_TDM 0
#endif

__device__ __forceinline__ v8f wmma_bf16(v16bf a, v16bf b, v8f c) {
  return __builtin_amdgcn_wmma_f32_16x16x32_bf16(false, a, false, b,
                                                 (short)0, c, false, false);
}

__device__ __forceinline__ unsigned short bf16_bits(float f) {
  __bf16 b = (__bf16)f;
  union { __bf16 h; unsigned short u; } cv;
  cv.h = b;
  return cv.u;
}

#if HAVE_TDM
// 1-D contiguous TDM copy: global (bf16 elements) -> LDS.
__device__ __forceinline__ void tdm_load_1d_bf16(unsigned lds_off,
                                                 const void* gptr,
                                                 unsigned nelem) {
  unsigned long long ga = (unsigned long long)gptr;
  u32x4 g0;
  g0[0] = 1u;                                        // count=1, user mode
  g0[1] = lds_off;                                   // lds_addr (bytes)
  g0[2] = (unsigned)(ga & 0xFFFFFFFFu);              // global_addr[31:0]
  g0[3] = (unsigned)((ga >> 32) & 0x01FFFFFFu)       // global_addr[56:32]
          | (2u << 30);                              // type = 2 ("image")
  i32x8 g1;
  g1[0] = (int)(1u << 16);                           // wg_mask=0, data_size=2B
  g1[1] = (int)((nelem & 0xFFFFu) << 16);            // tensor_dim0[15:0]
  g1[2] = (int)((nelem >> 16) & 0xFFFFu)             // tensor_dim0[31:16]
          | (int)(1u << 16);                         // tensor_dim1 = 1
  g1[3] = (int)((nelem & 0xFFFFu) << 16);            // tile_dim0
  g1[4] = 1;                                         // tile_dim1 = 1
  g1[5] = (int)nelem;                                // tensor_dim0_stride
  g1[6] = 0;
  g1[7] = 0;
  i32x4 z4 = {0, 0, 0, 0};
#if __clang_major__ >= 23
  i32x8 z8 = {0, 0, 0, 0, 0, 0, 0, 0};
  __builtin_amdgcn_tensor_load_to_lds(g0, g1, z4, z4, z8, 0);
#else
  __builtin_amdgcn_tensor_load_to_lds(g0, g1, z4, z4, 0);
#endif
}
#endif

#define DCAT 1152   // 9 * 128 concatenated GCN features
#define FCN  128
#define KPC  64     // PCA components

// ---------------------------------------------------------------------------
// Degree / normalization
// ---------------------------------------------------------------------------
__global__ void k_deginit(float* deg, int N) {
  int i = blockIdx.x * blockDim.x + threadIdx.x;
  if (i < N) deg[i] = 1.0f;                          // self-loop weight 1
}

__global__ void k_degedge(const int* __restrict__ dst,
                          const float* __restrict__ ew,
                          float* __restrict__ deg, int E) {
  int e = blockIdx.x * blockDim.x + threadIdx.x;
  if (e < E) atomicAdd(&deg[dst[e]], ew[e]);
}

__global__ void k_dinv(const float* __restrict__ deg, float* __restrict__ dinv,
                       int N) {
  int i = blockIdx.x * blockDim.x + threadIdx.x;
  if (i < N) {
    float d = deg[i];
    dinv[i] = (d > 0.f) ? rsqrtf(d) : 0.f;
  }
}

// ---------------------------------------------------------------------------
// bf16 conversions (global scratch)
// ---------------------------------------------------------------------------
__global__ void k_cvt_bf(const float* __restrict__ src,
                         unsigned short* __restrict__ dst, int n) {
  int i = blockIdx.x * blockDim.x + threadIdx.x;
  if (i < n) dst[i] = bf16_bits(src[i]);
}

// Wt[c][k] = bf16(W[k][c])  (128x128 transpose + convert)
__global__ void k_cvt_wt(const float* __restrict__ W,
                         unsigned short* __restrict__ Wt) {
  int i = blockIdx.x * blockDim.x + threadIdx.x;
  if (i >= FCN * FCN) return;
  int c = i >> 7, k = i & 127;
  Wt[i] = bf16_bits(W[k * FCN + c]);
}

// ---------------------------------------------------------------------------
// H = X @ W   (N x 128) @ (128 x 128), bf16 WMMA.  N is a multiple of 16.
// Block: 256 threads = 8 waves; 16 rows x 128 cols per block.
// X rows staged into LDS in the exact A-fragment permutation; W tile staged
// (pre-transposed bf16) via TDM so each fragment is one 32B LDS vector load.
// ---------------------------------------------------------------------------
__global__ __launch_bounds__(256) void k_gemm_xw(
    const float* __restrict__ X, long xstride,
    const unsigned short* __restrict__ Wtb,   // [c][k] bf16
    float* __restrict__ H) {
  __shared__ __align__(64) __bf16 sA[16 * FCN];   // permuted A rows
  __shared__ __align__(64) __bf16 sW[FCN * FCN];  // [c][k]
  const int rBase = blockIdx.x * 16;
  const int tid = threadIdx.x;

  // Stage X rows so lane-half h reads 16 contiguous bf16 per chunk:
  // within each 32-k block, position = q (k = 16h + s).
  for (int idx = tid; idx < 16 * FCN; idx += 256) {
    int r = idx >> 7, c = idx & 127;
    float v = X[(long)(rBase + r) * xstride + c];
    int kb = c >> 5, q = c & 31;
    int hh = (q & 15) >> 3;
    int pos = q;
    sA[r * FCN + kb * 32 + pos] = (__bf16)v;
  }
#if HAVE_TDM
  if (tid < 32) {                                  // wave 0 drives the TDM
    unsigned lds_off = (unsigned)(size_t)(void*)sW;
    tdm_load_1d_bf16(lds_off, Wtb, FCN * FCN);
    __builtin_amdgcn_s_wait_tensorcnt(0);
  }
#else
  {
    unsigned short* sWu = (unsigned short*)sW;
    for (int idx = tid; idx < FCN * FCN; idx += 256) sWu[idx] = Wtb[idx];
  }
#endif
  __syncthreads();

  const int wv = tid >> 5;                         // wave id = col tile
  const int l = tid & 31, ln = l & 15, h = l >> 4;
  const __bf16* sAp = sA + ln * FCN + h * 16;
  const __bf16* sWp = sW + (wv * 16 + ln) * FCN + h * 16;
  v8f acc = {};
#pragma unroll
  for (int kb = 0; kb < 4; ++kb) {                 // K = 128 in chunks of 32
    v16bf a = *(const v16bf*)(sAp + kb * 32);
    v16bf b = *(const v16bf*)(sWp + kb * 32);
    acc = wmma_bf16(a, b, acc);
  }
#pragma unroll
  for (int vv = 0; vv < 8; ++vv) {
    int r = rBase + vv + 8 * h;
    H[(long)r * FCN + wv * 16 + ln] = acc[vv];     // N % 16 == 0: no guard
  }
}

// ---------------------------------------------------------------------------
// out[r, colOff+f] = dinv[r]^2 * H[r,f] + bias[f]   (self loop + bias)
// ---------------------------------------------------------------------------
__global__ void k_initout(const float* __restrict__ H,
                          const float* __restrict__ dinv,
                          const float* __restrict__ bias,
                          float* __restrict__ Xcat, int colOff, int N) {
  int idx = blockIdx.x * blockDim.x + threadIdx.x;
  if (idx >= N * FCN) return;
  int r = idx >> 7, f = idx & 127;
  float dv = dinv[r];
  Xcat[(long)r * DCAT + colOff + f] = dv * dv * H[idx] + bias[f];
}

// ---------------------------------------------------------------------------
// Edge scatter: out[dst] += dinv[src]*ew*dinv[dst] * H[src].  32 lanes/edge.
// ---------------------------------------------------------------------------
__global__ void k_scatter(const int* __restrict__ src,
                          const int* __restrict__ dst,
                          const float* __restrict__ ew,
                          const float* __restrict__ dinv,
                          const float* __restrict__ H,
                          float* __restrict__ Xcat, int colOff, int E) {
  long tid = (long)blockIdx.x * blockDim.x + threadIdx.x;
  long e = tid >> 5;
  if (e >= E) return;
  int f4 = (int)(tid & 31) * 4;
  int s = src[e], d = dst[e];
  float coef = dinv[s] * ew[e] * dinv[d];
  const float4 hv = *(const float4*)(H + (long)s * FCN + f4);
  float* o = Xcat + (long)d * DCAT + colOff + f4;
  atomicAdd(o + 0, coef * hv.x);
  atomicAdd(o + 1, coef * hv.y);
  atomicAdd(o + 2, coef * hv.z);
  atomicAdd(o + 3, coef * hv.w);
}

__global__ void k_relu_slice(float* __restrict__ Xcat, int colOff, int N) {
  int idx = blockIdx.x * blockDim.x + threadIdx.x;
  if (idx >= N * FCN) return;
  int r = idx >> 7, f = idx & 127;
  float* p = Xcat + (long)r * DCAT + colOff + f;
  *p = fmaxf(*p, 0.f);
}

// ---------------------------------------------------------------------------
// Column means: zero + chunked partial sums (global atomics) + divide
// ---------------------------------------------------------------------------
__global__ void k_zero_f(float* __restrict__ p, int n) {
  int i = blockIdx.x * blockDim.x + threadIdx.x;
  if (i < n) p[i] = 0.f;
}

#define MEAN_CHUNK 1024
__global__ void k_colmean_part(const float* __restrict__ X,
                               float* __restrict__ msum, int N, int nchunk) {
  int idx = blockIdx.x * blockDim.x + threadIdx.x;
  if (idx >= DCAT * nchunk) return;
  int j = idx % DCAT, c = idx / DCAT;
  int r0 = c * MEAN_CHUNK;
  int r1 = min(r0 + MEAN_CHUNK, N);
  float s = 0.f;
  for (int r = r0; r < r1; ++r) s += X[(long)r * DCAT + j];
  atomicAdd(&msum[j], s);
}

__global__ void k_mean_div(float* __restrict__ mean, int N) {
  int j = blockIdx.x * blockDim.x + threadIdx.x;
  if (j < DCAT) mean[j] /= (float)N;
}

// Centered bf16 copy: Xcb[r][j] = bf16(X[r][j] - mean[j])
__global__ void k_center_bf(const float* __restrict__ X,
                            const float* __restrict__ mean,
                            unsigned short* __restrict__ Xcb, long n) {
  long i = (long)blockIdx.x * blockDim.x + threadIdx.x;
  if (i >= n) return;
  int j = (int)(i % DCAT);
  Xcb[i] = bf16_bits(X[i] - mean[j]);
}

// ---------------------------------------------------------------------------
// Gram = Xc^T Xc  (1152 x 1152) from the centered bf16 copy; one wave per
// 16x16 output tile.
// ---------------------------------------------------------------------------
__global__ __launch_bounds__(256) void k_gram(
    const unsigned short* __restrict__ Xb, int N, float* __restrict__ G) {
  const int T = DCAT / 16;                         // 72
  int wave = blockIdx.x * 8 + (threadIdx.x >> 5);
  if (wave >= T * T) return;
  int ti = wave / T, tj = wave % T;
  int l = threadIdx.x & 31, ln = l & 15, h = l >> 4;
  const __bf16* X = (const __bf16*)Xb;
  v8f acc = {};
  for (int r = 0; r < N; r += 32) {
    if (r + 32 < N) {
      __builtin_prefetch(&Xb[(long)(r + 32) * DCAT + ti * 16 + ln], 0, 3);
      __builtin_prefetch(&Xb[(long)(r + 32) * DCAT + tj * 16 + ln], 0, 3);
    }
    v16bf a, b;
#pragma unroll
    for (int e = 0; e < 8; ++e) {
      a[e]     = X[(long)(r + h * 16 + e) * DCAT + ti * 16 + ln];
      a[8 + e] = X[(long)(r + h * 16 + 8 + e) * DCAT + ti * 16 + ln];
    }
#pragma unroll
    for (int e = 0; e < 16; ++e)
      b[e] = X[(long)(r + h * 16 + e) * DCAT + tj * 16 + ln];
    acc = wmma_bf16(a, b, acc);
  }
#pragma unroll
  for (int vv = 0; vv < 8; ++vv) {
    int i = ti * 16 + vv + 8 * h, j = tj * 16 + ln;
    G[(long)i * DCAT + j] = acc[vv];
  }
}

// cov = G / (N-1), in place (G already centered)
__global__ void k_cov(float* __restrict__ G, int N) {
  int idx = blockIdx.x * blockDim.x + threadIdx.x;
  if (idx >= DCAT * DCAT) return;
  G[idx] /= (float)(N - 1);
}

// Deterministic pseudo-random init for orthogonal iteration
__global__ void k_yinit(float* __restrict__ Y) {
  int idx = blockIdx.x * blockDim.x + threadIdx.x;
  if (idx >= DCAT * KPC) return;
  unsigned u = (unsigned)idx * 2654435761u + 12345u;
  u ^= u >> 16; u *= 2246822519u; u ^= u >> 13;
  Y[idx] = ((float)(u & 0xFFFF) / 32768.0f) - 1.0f;
}

// ---------------------------------------------------------------------------
// Z = cov @ Y   (1152x1152)@(1152x64), WMMA; 288 waves.
// ---------------------------------------------------------------------------
__global__ __launch_bounds__(256) void k_gemm_ay(const float* __restrict__ A,
                                                 const float* __restrict__ Y,
                                                 float* __restrict__ Z) {
  int wave = blockIdx.x * 8 + (threadIdx.x >> 5);
  if (wave >= (DCAT / 16) * (KPC / 16)) return;
  int ti = wave >> 2, tj = wave & 3;
  int l = threadIdx.x & 31, ln = l & 15, h = l >> 4;
  v8f acc = {};
  for (int kb = 0; kb < DCAT; kb += 32) {
    v16bf a, b;
#pragma unroll
    for (int e = 0; e < 8; ++e) {
      a[e]     = (__bf16)A[(long)(ti * 16 + ln) * DCAT + kb + h * 16 + e];
      a[8 + e] = (__bf16)A[(long)(ti * 16 + ln) * DCAT + kb + h * 16 + 8 + e];
    }
#pragma unroll
    for (int e = 0; e < 16; ++e)
      b[e] = (__bf16)Y[(long)(kb + h * 16 + e) * KPC + tj * 16 + ln];
    acc = wmma_bf16(a, b, acc);
  }
#pragma unroll
  for (int vv = 0; vv < 8; ++vv)
    Z[(long)(ti * 16 + vv + 8 * h) * KPC + tj * 16 + ln] = acc[vv];
}

// ---------------------------------------------------------------------------
// Single-block modified Gram-Schmidt QR on A [1152 x 64] (in place).
// Column dots via LDS float atomics (ds_add_f32).
// ---------------------------------------------------------------------------
__global__ __launch_bounds__(1024) void k_qr(float* __restrict__ A) {
  __shared__ float red[1024];
  __shared__ float dots[KPC];
  __shared__ float snorm;
  const int R = DCAT, C = KPC;
  const int tid = threadIdx.x, nt = blockDim.x;
  for (int j = 0; j < C; ++j) {
    float s = 0.f;
    for (int r = tid; r < R; r += nt) { float v = A[r * C + j]; s += v * v; }
    red[tid] = s;
    __syncthreads();
    for (int off = nt >> 1; off > 0; off >>= 1) {
      if (tid < off) red[tid] += red[tid + off];
      __syncthreads();
    }
    if (tid == 0) snorm = rsqrtf(fmaxf(red[0], 1e-24f));
    __syncthreads();
    float sc = snorm;
    for (int r = tid; r < R; r += nt) A[r * C + j] *= sc;
    if (tid < C) dots[tid] = 0.f;
    __threadfence_block();
    __syncthreads();
    for (int idx = tid; idx < R * C; idx += nt) {
      int r = idx >> 6, k = idx & 63;
      if (k > j) atomicAdd(&dots[k], A[r * C + j] * A[r * C + k]);
    }
    __syncthreads();
    for (int idx = tid; idx < R * C; idx += nt) {
      int r = idx >> 6, k = idx & 63;
      if (k > j) A[r * C + k] -= dots[k] * A[r * C + j];
    }
    __threadfence_block();
    __syncthreads();
  }
}

// ---------------------------------------------------------------------------
// O = Xc @ V   (N x 1152)@(1152 x 64) from centered bf16 copy; A fragments
// are pure 16B global vector loads.  N % 16 == 0.
// ---------------------------------------------------------------------------
__global__ __launch_bounds__(256) void k_project(
    const unsigned short* __restrict__ Xb, const float* __restrict__ V,
    float* __restrict__ O, int N) {
  int wave = blockIdx.x * 8 + (threadIdx.x >> 5);
  int nwaves = (N / 16) * (KPC / 16);
  if (wave >= nwaves) return;
  int tr = wave >> 2, tc = wave & 3;
  int l = threadIdx.x & 31, ln = l & 15, h = l >> 4;
  const __bf16* Xp = (const __bf16*)Xb + (long)(tr * 16 + ln) * DCAT + h * 16;
  v8f acc = {};
  for (int kb = 0; kb < DCAT; kb += 32) {
    v8bf alo = *(const v8bf*)(Xp + kb);
    v8bf ahi = *(const v8bf*)(Xp + kb + 8);
    v16bf a;
#pragma unroll
    for (int e = 0; e < 8; ++e) { a[e] = alo[e]; a[8 + e] = ahi[e]; }
    v16bf b;
#pragma unroll
    for (int e = 0; e < 16; ++e)
      b[e] = (__bf16)V[(long)(kb + h * 16 + e) * KPC + tc * 16 + ln];
    acc = wmma_bf16(a, b, acc);
  }
#pragma unroll
  for (int vv = 0; vv < 8; ++vv)
    O[(long)(tr * 16 + vv + 8 * h) * KPC + tc * 16 + ln] = acc[vv];
}

// ---------------------------------------------------------------------------
// Row l2-normalize the 64 PCA cols, concat with 64 graph-embedding cols.
// ---------------------------------------------------------------------------
__global__ void k_l2cat(const float* __restrict__ Pm,
                        const float* __restrict__ Emb, float* __restrict__ O,
                        int N) {
  int r = blockIdx.x * blockDim.x + threadIdx.x;
  if (r >= N) return;
  float s = 0.f;
  for (int c = 0; c < KPC; ++c) { float v = Pm[(long)r * KPC + c]; s += v * v; }
  float inv = 1.f / fmaxf(sqrtf(s), 1e-12f);
  for (int c = 0; c < KPC; ++c) O[(long)r * FCN + c] = Pm[(long)r * KPC + c] * inv;
  for (int c = 0; c < KPC; ++c) O[(long)r * FCN + KPC + c] = Emb[(long)r * KPC + c];
}

// ---------------------------------------------------------------------------
// Fused bilinear decoder: basis[p,k] = c_p^T W_k d_p (WMMA tiles of 16 pairs),
// out = relu(basis @ dec_cls). All operands pre-converted to bf16 in global
// memory so fragments are pure vector loads; cross-lane shfl_xor fold.
// ---------------------------------------------------------------------------
__global__ __launch_bounds__(256) void k_dec(
    const unsigned short* __restrict__ Cb,    // [NC x 128] bf16
    const unsigned short* __restrict__ Db,    // [ND x 128] bf16
    const int* __restrict__ ci, const int* __restrict__ di,
    const unsigned short* __restrict__ Wb,    // [2 x 128 x 128] bf16
    const float* __restrict__ cls, float* __restrict__ out, int P) {
  int wave = blockIdx.x * 8 + (threadIdx.x >> 5);
  if (wave >= P / 16) return;
  int pb = wave * 16;
  int l = threadIdx.x & 31, ln = l & 15, h = l >> 4;
  const __bf16* Dp = (const __bf16*)Db + (long)di[pb + ln] * FCN;  // m = ln
  const __bf16* Cp = (const __bf16*)Cb;
  int cidx[8];
#pragma unroll
  for (int vv = 0; vv < 8; ++vv) cidx[vv] = ci[pb + vv + 8 * h];

  float basis[2][8];
#pragma unroll
  for (int k = 0; k < 2; ++k)
#pragma unroll
    for (int vv = 0; vv < 8; ++vv) basis[k][vv] = 0.f;

  for (int k = 0; k < 2; ++k) {
    const __bf16* Wk = (const __bf16*)Wb + (long)k * FCN * FCN;
#pragma unroll
    for (int it = 0; it < 8; ++it) {       // i-tiles of U = D @ Wk^T
      v8f acc = {};
#pragma unroll
      for (int jb = 0; jb < 4; ++jb) {     // K = 128 over j
        v8bf alo = *(const v8bf*)(Dp + jb * 32 + h * 16);
        v8bf ahi = *(const v8bf*)(Dp + jb * 32 + h * 16 + 8);
        v16bf a;
#pragma unroll
        for (int e = 0; e < 8; ++e) { a[e] = alo[e]; a[8 + e] = ahi[e]; }
        // B[kk][n] = Wk[i = it*16+n][j = jb*32+kk]; contiguous in kk
        v16bf b = *(const v16bf*)(Wk + (long)(it * 16 + ln) * FCN + jb * 32 +
                                  h * 16);
        acc = wmma_bf16(a, b, acc);
      }
#pragma unroll
      for (int vv = 0; vv < 8; ++vv) {
        float cv = (float)Cp[(long)cidx[vv] * FCN + it * 16 + ln];
        basis[k][vv] += acc[vv] * cv;      // partial sum over i for this n=ln
      }
    }
  }
  // Reduce over the 16 lanes of each half-wave (sum over n)
#pragma unroll
  for (int k = 0; k < 2; ++k)
#pragma unroll
    for (int vv = 0; vv < 8; ++vv) {
      float x = basis[k][vv];
      for (int off = 8; off > 0; off >>= 1) x += __shfl_xor(x, off, 16);
      basis[k][vv] = x;                    // valid at ln == 0
    }
  if (ln == 0) {
    float c00 = cls[0], c01 = cls[1], c10 = cls[2], c11 = cls[3];
#pragma unroll
    for (int vv = 0; vv < 8; ++vv) {
      int m = vv + 8 * h;
      float b0 = basis[0][vv], b1 = basis[1][vv];
      out[(long)(pb + m) * 2 + 0] = fmaxf(b0 * c00 + b1 * c10, 0.f);
      out[(long)(pb + m) * 2 + 1] = fmaxf(b0 * c01 + b1 * c11, 0.f);
    }
  }
}

// ---------------------------------------------------------------------------
// Host orchestration
// ---------------------------------------------------------------------------
extern "C" void kernel_launch(void* const* d_in, const int* in_sizes, int n_in,
                              void* d_out, int out_size, void* d_ws,
                              size_t ws_size, hipStream_t stream) {
  (void)in_sizes; (void)n_in; (void)out_size; (void)ws_size;
  const int NC = 60000, ND = 20000, EC = 1920000, ED = 640000, P = 200000;

  const int*   eC[3] = {(const int*)d_in[0], (const int*)d_in[2], (const int*)d_in[4]};
  const float* wC[3] = {(const float*)d_in[1], (const float*)d_in[3], (const float*)d_in[5]};
  const int*   eD[3] = {(const int*)d_in[6], (const int*)d_in[8], (const int*)d_in[10]};
  const float* wD[3] = {(const float*)d_in[7], (const float*)d_in[9], (const float*)d_in[11]};
  const float* x_fc = (const float*)d_in[12];
  const float* y_fd = (const float*)d_in[13];
  const float* embC = (const float*)d_in[14];
  const float* embD = (const float*)d_in[15];
  const int*   ci   = (const int*)d_in[16];
  const int*   di   = (const int*)d_in[17];
  const float* Wc   = (const float*)d_in[18];
  const float* bc   = (const float*)d_in[19];
  const float* Wd   = (const float*)d_in[20];
  const float* bd   = (const float*)d_in[21];
  const float* decW = (const float*)d_in[22];
  const float* cls  = (const float*)d_in[23];

  char* base = (char*)d_ws;
  size_t off = 0;
  auto carveB = [&](size_t bytes) {
    void* p = base + off;
    off += (bytes + 255) & ~(size_t)255;
    return p;
  };
  float* XMcat = (float*)carveB((size_t)NC * DCAT * 4);
  float* YDcat = (float*)carveB((size_t)ND * DCAT * 4);
  float* Hbuf  = (float*)carveB((size_t)NC * FCN * 4);
  float* degb  = (float*)carveB((size_t)NC * 4);
  float* dinvb = (float*)carveB((size_t)NC * 4);
  float* meanv = (float*)carveB((size_t)DCAT * 4);
  float* gram  = (float*)carveB((size_t)DCAT * DCAT * 4);
  float* Ybuf  = (float*)carveB((size_t)DCAT * KPC * 4);
  float* Ynew  = (float*)carveB((size_t)DCAT * KPC * 4);
  float* Pbuf  = (float*)carveB((size_t)NC * KPC * 4);
  float* XM2   = (float*)carveB((size_t)NC * FCN * 4);
  float* YD2   = (float*)carveB((size_t)ND * FCN * 4);
  unsigned short* Wtb   = (unsigned short*)carveB((size_t)FCN * FCN * 2);
  unsigned short* decWb = (unsigned short*)carveB((size_t)2 * FCN * FCN * 2);
  unsigned short* XM2b  = (unsigned short*)carveB((size_t)NC * FCN * 2);
  unsigned short* YD2b  = (unsigned short*)carveB((size_t)ND * FCN * 2);
  unsigned short* Xcb   = (unsigned short*)carveB((size_t)NC * DCAT * 2);

  auto run_side = [&](const float* x0, const float* W9, const float* b9,
                      const int* const* eArr, const float* const* wArr, int N,
                      int E, float* Xcat) {
    for (int g = 0; g < 3; ++g) {
      const int* src = eArr[g];
      const int* dst = eArr[g] + E;
      const float* ew = wArr[g];
      k_deginit<<<(N + 255) / 256, 256, 0, stream>>>(degb, N);
      k_degedge<<<(E + 255) / 256, 256, 0, stream>>>(dst, ew, degb, E);
      k_dinv<<<(N + 255) / 256, 256, 0, stream>>>(degb, dinvb, N);
      const float* xin = x0;
      long xstride = FCN;
      for (int layer = 0; layer < 3; ++layer) {
        const float* W = W9 + (size_t)(g * 3 + layer) * FCN * FCN;
        const float* b = b9 + (size_t)(g * 3 + layer) * FCN;
        int colOff = (g * 3 + layer) * FCN;
        k_cvt_wt<<<(FCN * FCN + 255) / 256, 256, 0, stream>>>(W, Wtb);
        k_gemm_xw<<<N / 16, 256, 0, stream>>>(xin, xstride, Wtb, Hbuf);
        k_initout<<<(N * FCN + 255) / 256, 256, 0, stream>>>(Hbuf, dinvb, b,
                                                             Xcat, colOff, N);
        long sthreads = (long)E * 32;
        k_scatter<<<(unsigned)((sthreads + 255) / 256), 256, 0, stream>>>(
            src, dst, ew, dinvb, Hbuf, Xcat, colOff, E);
        k_relu_slice<<<(N * FCN + 255) / 256, 256, 0, stream>>>(Xcat, colOff, N);
        xin = Xcat + colOff;
        xstride = DCAT;
      }
    }
  };

  auto pca_side = [&](float* Xcat, int N, const float* Emb, float* X2,
                      unsigned short* X2b) {
    // column means (chunked atomics), centered bf16 copy
    int nchunk = (N + MEAN_CHUNK - 1) / MEAN_CHUNK;
    k_zero_f<<<(DCAT + 255) / 256, 256, 0, stream>>>(meanv, DCAT);
    k_colmean_part<<<(DCAT * nchunk + 255) / 256, 256, 0, stream>>>(Xcat, meanv,
                                                                    N, nchunk);
    k_mean_div<<<(DCAT + 255) / 256, 256, 0, stream>>>(meanv, N);
    long nXc = (long)N * DCAT;
    k_center_bf<<<(unsigned)((nXc + 255) / 256), 256, 0, stream>>>(Xcat, meanv,
                                                                   Xcb, nXc);
    const int T = DCAT / 16;
    k_gram<<<(T * T + 7) / 8, 256, 0, stream>>>(Xcb, N, gram);
    k_cov<<<(DCAT * DCAT + 255) / 256, 256, 0, stream>>>(gram, N);
    k_yinit<<<(DCAT * KPC + 255) / 256, 256, 0, stream>>>(Ybuf);
    k_qr<<<1, 1024, 0, stream>>>(Ybuf);
    float* Ya = Ybuf;
    float* Yb = Ynew;
    for (int it = 0; it < 16; ++it) {     // orthogonal iteration ~ top-64 eigh
      k_gemm_ay<<<((DCAT / 16) * (KPC / 16) + 7) / 8, 256, 0, stream>>>(gram,
                                                                        Ya, Yb);
      k_qr<<<1, 1024, 0, stream>>>(Yb);
      float* t = Ya; Ya = Yb; Yb = t;
    }
    int nwaves = (N / 16) * (KPC / 16);
    k_project<<<(nwaves + 7) / 8, 256, 0, stream>>>(Xcb, Ya, Pbuf, N);
    k_l2cat<<<(N + 255) / 256, 256, 0, stream>>>(Pbuf, Emb, X2, N);
    k_cvt_bf<<<(N * FCN + 255) / 256, 256, 0, stream>>>(X2, X2b, N * FCN);
  };

  run_side(x_fc, Wc, bc, eC, wC, NC, EC, XMcat);
  run_side(y_fd, Wd, bd, eD, wD, ND, ED, YDcat);
  pca_side(XMcat, NC, embC, XM2, XM2b);
  pca_side(YDcat, ND, embD, YD2, YD2b);

  k_cvt_bf<<<(2 * FCN * FCN + 255) / 256, 256, 0, stream>>>(decW, decWb,
                                                            2 * FCN * FCN);
  k_dec<<<(P / 16 + 7) / 8, 256, 0, stream>>>(XM2b, YD2b, ci, di, decWb, cls,
                                              (float*)d_out, P);
}
